// Attention_76794015252633
// MI455X (gfx1250) — matrix operand order repaired and checked by reading, not compiled, not run on hardware
//
#include <hip/hip_runtime.h>

#define BB 32
#define TT 1024
#define SS 1024
#define DD 512

typedef __bf16 bf16_t;
typedef __attribute__((ext_vector_type(16))) __bf16 v16bf;
typedef __attribute__((ext_vector_type(8)))  __bf16 v8bf;
typedef __attribute__((ext_vector_type(8)))  float  v8f;
typedef __attribute__((ext_vector_type(4)))  float  f32x4;

static __device__ __forceinline__ __bf16 f2bf(float f) { return (__bf16)f; }
static __device__ __forceinline__ float  bf2f(__bf16 b) { return (float)b; }

static __device__ __forceinline__ v8f wmma_bf16(v16bf a, v16bf b, v8f c) {
  // D = A(16x32 bf16) * B(32x16 bf16) + C(16x16 f32)
  return __builtin_amdgcn_wmma_f32_16x16x32_bf16(false, a, false, b, (short)0, c,
                                                 false, false);
}

static __device__ __forceinline__ v16bf cat8(v8bf lo, v8bf hi) {
  v16bf r;
#pragma unroll
  for (int i = 0; i < 8; ++i) {
    r[i] = lo[i];
    r[i + 8] = hi[i];
  }
  return r;
}

// ---------------------------------------------------------------------------
// Stage 1a: Abf[b][s][d] = bf16(attn[b][s][d])      (row-major copy)
//           Vt [b][d][s] = bf16(attn[b][s][d])      (transpose)
// ---------------------------------------------------------------------------
__global__ __launch_bounds__(256) void attn_convert_kernel(
    const float* __restrict__ attn, bf16_t* __restrict__ Abf,
    bf16_t* __restrict__ Vt) {
  size_t i = (size_t)blockIdx.x * 256 + threadIdx.x;  // over B*S*D = 2^24
  int d = (int)(i & (DD - 1));
  int s = (int)((i >> 9) & (SS - 1));
  int b = (int)(i >> 19);
  bf16_t v = f2bf(attn[i]);
  Abf[i] = v;                                   // coalesced
  Vt[((size_t)b * DD + d) * SS + s] = v;        // scattered (one-time cost)
}

// Stage 1b: Wbf = bf16(W), D*D = 2^18 elements
__global__ __launch_bounds__(256) void w_convert_kernel(
    const float* __restrict__ W, bf16_t* __restrict__ Wbf) {
  size_t i = (size_t)blockIdx.x * 256 + threadIdx.x;
  Wbf[i] = f2bf(W[i]);
}

// ---------------------------------------------------------------------------
// Stage 1c: Kp[b][s][e] = bf16( sum_d attn[b][s][d] * W[e][d] )  via WMMA.
// Pure-bf16 operand feeds: 4 vector loads + 1 WMMA per K-step.
// ---------------------------------------------------------------------------
__global__ __launch_bounds__(128) void proj_gemm_kernel(
    const bf16_t* __restrict__ Abf, const bf16_t* __restrict__ Wbf,
    bf16_t* __restrict__ Kp) {
  const int lane = threadIdx.x & 31;
  const int lm = lane & 15, lh = lane >> 4;
  const int wid = blockIdx.x * 4 + (threadIdx.x >> 5);
  const int r0 = (wid >> 5) * 16;  // row tile over B*S
  const int c0 = (wid & 31) * 16;  // col tile over D
  const bf16_t* arow = Abf + (size_t)(r0 + lm) * DD + lh * 8;
  const bf16_t* wrow = Wbf + (size_t)(c0 + lm) * DD + lh * 8;
  v8f acc = {};
#pragma unroll
  for (int kc = 0; kc < 16; ++kc) {
    // A layout: lane=M, e<8 -> K = kc*32 + lh*8 + e ; e>=8 -> +16
    const bf16_t* ga = arow + kc * 32;
    v16bf av = cat8(*(const v8bf*)(ga), *(const v8bf*)(ga + 16));
    // B layout: lane=N, e<8 -> K = kc*32 + lh*8 + e ; e>=8 -> +16 ; B[k][n] = W[n][k]
    v16bf bv = cat8(*(const v8bf*)(wrow + kc * 32), *(const v8bf*)(wrow + kc * 32 + 16));
    acc = wmma_bf16(av, bv, acc);
  }
  // C layout: lane=N (col c0+lm), VGPR v -> row r0 + v + lh*8
#pragma unroll
  for (int v = 0; v < 8; ++v)
    Kp[(size_t)(r0 + v + lh * 8) * DD + c0 + lm] = f2bf(acc[v]);
}

// ---------------------------------------------------------------------------
// Stage 2: fused  scores -> softmax (in LDS) -> P @ V
// One wave per 16-row T tile; 2 waves per block; 32 KB LDS per wave for P.
// ---------------------------------------------------------------------------
__global__ __launch_bounds__(64) void attn_fused_kernel(
    const float* __restrict__ mainx, const bf16_t* __restrict__ Kp,
    const bf16_t* __restrict__ Vt, float* __restrict__ out) {
  extern __shared__ char smem[];
  const int lane = threadIdx.x & 31;
  const int wave = threadIdx.x >> 5;
  const int lm = lane & 15, lh = lane >> 4;
  const int wtile = blockIdx.x * 2 + wave;  // 0..2047
  const int b = wtile >> 6;
  const int t0 = (wtile & 63) << 4;

  bf16_t* P = (bf16_t*)smem + wave * 16 * SS;  // [16][1024] bf16 stripe

  // ---- preload Q tile (16 x 512) into A-layout registers, cvt f32->bf16
  v16bf q[16];
  const float* qrow = mainx + ((size_t)b * TT + t0 + lm) * DD + lh * 8;
#pragma unroll
  for (int kc = 0; kc < 16; ++kc) {
    const float* g = qrow + kc * 32;
    f32x4 a0 = *(const f32x4*)(g);
    f32x4 a1 = *(const f32x4*)(g + 4);
    f32x4 a2 = *(const f32x4*)(g + 16);
    f32x4 a3 = *(const f32x4*)(g + 20);
    v16bf av;
#pragma unroll
    for (int i = 0; i < 4; ++i) {
      av[i]      = f2bf(a0[i]);
      av[i + 4]  = f2bf(a1[i]);
      av[i + 8]  = f2bf(a2[i]);
      av[i + 12] = f2bf(a3[i]);
    }
    q[kc] = av;
  }

  // ---- scores pass: 64 tiles of 16 S-columns, 16 WMMA each over D=512
  const bf16_t* krow0 = Kp + ((size_t)b * SS + lm) * DD + lh * 8;
  for (int st = 0; st < 64; ++st) {
    const bf16_t* krow = krow0 + (size_t)(st * 16) * DD;
    // prefetch next S-tile's K rows (global_prefetch_b8)
    __builtin_prefetch(krow + (size_t)16 * DD, 0, 0);
    v8f acc = {};
#pragma unroll
    for (int kc = 0; kc < 16; ++kc) {
      v16bf bv = cat8(*(const v8bf*)(krow + kc * 32), *(const v8bf*)(krow + kc * 32 + 16));
      acc = wmma_bf16(q[kc], bv, acc);
    }
    const int col = st * 16 + lm;
#pragma unroll
    for (int v = 0; v < 8; ++v) P[(v + lh * 8) * SS + col] = f2bf(acc[v]);
  }

  // ---- softmax over S (wave-local; lanes r and r+16 split row r's columns)
  {
    bf16_t* prow = P + lm * SS + lh * 512;
    float m = -3.0e38f;
    for (int c = 0; c < 512; c += 8) {
      v8bf x = *(const v8bf*)(prow + c);
#pragma unroll
      for (int i = 0; i < 8; ++i) m = fmaxf(m, bf2f(x[i]));
    }
    m = fmaxf(m, __shfl_xor(m, 16, 32));
    float sum = 0.0f;
    for (int c = 0; c < 512; c += 8) {
      v8bf x = *(const v8bf*)(prow + c);
#pragma unroll
      for (int i = 0; i < 8; ++i) sum += __expf(bf2f(x[i]) - m);
    }
    sum += __shfl_xor(sum, 16, 32);
    const float inv = 1.0f / sum;
    for (int c = 0; c < 512; c += 8) {
      v8bf x = *(const v8bf*)(prow + c);
      v8bf y;
#pragma unroll
      for (int i = 0; i < 8; ++i) y[i] = f2bf(__expf(bf2f(x[i]) - m) * inv);
      *(v8bf*)(prow + c) = y;
    }
  }

  // ---- PV pass: out[16 x 512] in chunks of 64 cols, K-loop over S=1024
  const bf16_t* vb = Vt + (size_t)b * DD * SS;
  float* ob = out + ((size_t)b * TT + t0) * DD;
  for (int dc = 0; dc < 8; ++dc) {
    v8f acc[4] = {};
    const bf16_t* vrow0 = vb + (size_t)(dc * 64 + lm) * SS + lh * 8;
    for (int st = 0; st < 32; ++st) {
      // A = P tile (16 x 32) from LDS: two 16B ds reads per lane
      const bf16_t* pr = P + lm * SS + st * 32 + lh * 8;
      v16bf a = cat8(*(const v8bf*)(pr), *(const v8bf*)(pr + 16));
      const bf16_t* vr = vrow0 + st * 32;
#pragma unroll
      for (int nt = 0; nt < 4; ++nt) {
        v16bf bv = cat8(*(const v8bf*)(vr + (size_t)nt * 16 * SS), *(const v8bf*)(vr + (size_t)nt * 16 * SS + 16));
        acc[nt] = wmma_bf16(a, bv, acc[nt]);
      }
    }
#pragma unroll
    for (int nt = 0; nt < 4; ++nt)
#pragma unroll
      for (int v = 0; v < 8; ++v)
        ob[(size_t)(v + lh * 8) * DD + dc * 64 + nt * 16 + lm] = acc[nt][v];
  }
}

// ---------------------------------------------------------------------------
extern "C" void kernel_launch(void* const* d_in, const int* in_sizes, int n_in,
                              void* d_out, int out_size, void* d_ws,
                              size_t ws_size, hipStream_t stream) {
  const float* mainx = (const float*)d_in[0];  // [B,T,D] f32
  const float* attn  = (const float*)d_in[1];  // [B,S,D] f32
  const float* W     = (const float*)d_in[2];  // [D,D]   f32
  float* out = (float*)d_out;                  // [B,T,D] f32

  bf16_t* Kp  = (bf16_t*)d_ws;                 // 32 MB: proj bf16 [B,S,D]
  bf16_t* Vt  = Kp + (size_t)BB * SS * DD;     // 32 MB: attn bf16 [B,D,S]
  bf16_t* Abf = Vt + (size_t)BB * DD * SS;     // 32 MB: attn bf16 [B,S,D]
  bf16_t* Wbf = Abf + (size_t)BB * SS * DD;    // 0.5 MB: W bf16 [D,D]

  // B*S*D = 2^24 elements
  attn_convert_kernel<<<(BB * SS * DD) / 256, 256, 0, stream>>>(attn, Abf, Vt);
  // D*D = 2^18 elements
  w_convert_kernel<<<(DD * DD) / 256, 256, 0, stream>>>(W, Wbf);
  // (B*S/16)*(D/16) = 65536 wave tiles, 4 waves/block
  proj_gemm_kernel<<<65536 / 4, 128, 0, stream>>>(Abf, Wbf, Kp);
  // B*T/16 = 2048 wave tiles, 2 waves/block, 64 KB dynamic LDS
  attn_fused_kernel<<<2048 / 2, 64, 2 * 16 * SS * (int)sizeof(bf16_t),
                      stream>>>(mainx, Kp, Vt, out);
}
